// ProfessionalWindowAttention_56573309222965
// MI455X (gfx1250) — matrix operand order repaired and checked by reading, not compiled, not run on hardware
//
#include <hip/hip_runtime.h>

typedef _Float16 half_t;
typedef __attribute__((ext_vector_type(4)))  _Float16 v4h;
typedef __attribute__((ext_vector_type(8)))  _Float16 v8h;
typedef __attribute__((ext_vector_type(16))) _Float16 v16h;
typedef __attribute__((ext_vector_type(8)))  float    v8f;

#define NH_    8
#define DIM_   256
#define HD_    32
#define NTOK   64
#define SHIFT_ 4
#define HW_    256
#define NWIN   2048
#define SCALE_ 0.17677669529663687f

// LDS pitches (half elements). All row strides are multiples of 8 halves (16B)
// so v8h accesses stay 16B-aligned.
#define XP 264
#define QP 40
#define PP 72
#define VP 72
#define QK_HEAD (2 * NTOK * QP)       // 5120 halves per head (q then k)
#define XS_SZ   (NTOK * XP)           // 16896
#define QK_OFF  XS_SZ
#define QK_SZ   (NH_ * QK_HEAD)       // 40960
#define VT_OFF  (QK_OFF + QK_SZ)
#define VT_HEAD (HD_ * VP)            // 2304
#define VT_SZ   (NH_ * VT_HEAD)       // 18432
#define SMEM_TOT (VT_OFF + VT_SZ)     // 76288 halves = 152,576 B

static __device__ inline v16h mk16(v8h lo, v8h hi) {
  v16h r;
#pragma unroll
  for (int i = 0; i < 8; ++i) { r[i] = lo[i]; r[i + 8] = hi[i]; }
  return r;
}

// A-matrix (16x32 f16) tile fragment. p = &src[row*pitch + k0], row-major.
// Lane layout (ISA 7.12.2): e<8 -> K = g*8+e ; e>=8 -> K = 16+g*8+(e-8).
static __device__ inline v16h loadA(const half_t* p, int g) {
  v8h lo = *(const v8h*)(p + g * 8);
  v8h hi = *(const v8h*)(p + 16 + g * 8);
  return mk16(lo, hi);
}

// B-matrix (32x16 f16) tile fragment. Source is "n-major": p = &src[n*pitch + k0],
// lane holds column n = lane&15, e<8 -> K = g*8+e ; e>=8 -> K = 16+g*8+(e-8).
static __device__ inline v16h loadB(const half_t* p, int g) {
  v8h lo = *(const v8h*)(p + g * 8);
  v8h hi = *(const v8h*)(p + 16 + g * 8);
  return mk16(lo, hi);
}

static __device__ inline v8f wmma_f16(v16h a, v16h b, v8f c) {
  return __builtin_amdgcn_wmma_f32_16x16x32_f16(false, a, false, b, (short)0, c,
                                                false, false);
}

// ---- prepack: f32 weights -> f16 (B-friendly row-major), bias table gather ----
__global__ void swin_prepack(const float* __restrict__ qkv_w,
                             const float* __restrict__ proj_w,
                             const float* __restrict__ bias_table,
                             const int* __restrict__ rel_index,
                             half_t* __restrict__ wq16, half_t* __restrict__ wp16,
                             float* __restrict__ bfull) {
  int id = blockIdx.x * 256 + threadIdx.x;
  if (id < 3 * DIM_ * DIM_) wq16[id] = (half_t)qkv_w[id];
  if (id < DIM_ * DIM_)     wp16[id] = (half_t)proj_w[id];
  if (id < NH_ * NTOK * NTOK) {
    int h = id >> 12;          // [h][m][n]
    int mn = id & 4095;
    bfull[id] = bias_table[rel_index[mn] * NH_ + h];
  }
}

// ---- fused shifted-window attention: 1 workgroup = 1 window, 1 wave = 1 head ----
__global__ __launch_bounds__(256, 1)
void swin_attn(const float* __restrict__ x, const float* __restrict__ qkv_b,
               const float* __restrict__ proj_b,
               const half_t* __restrict__ wq16, const half_t* __restrict__ wp16,
               const float* __restrict__ bfull, float* __restrict__ out) {
  __shared__ half_t sm[SMEM_TOT];

  const int tid  = threadIdx.x;
  const int w    = blockIdx.x;
  const int wb   = w >> 10;        // batch
  const int wi   = (w >> 5) & 31;  // window row
  const int wj   = w & 31;         // window col
  const int lane = tid & 31;
  const int h    = tid >> 5;       // wave id == head
  const int g    = (lane >> 4) & 1;
  const int l15  = lane & 15;

  half_t* xs = sm;                                // [64][XP]; reused as proj input later
  half_t* qs = sm + QK_OFF + h * QK_HEAD;         // [64][QP]
  half_t* ks = qs + NTOK * QP;                    // [64][QP]
  half_t* ps = sm + QK_OFF + h * QK_HEAD;         // alias over q/k: [64][PP]
  half_t* vt = sm + VT_OFF + h * VT_HEAD;         // [32][VP]  (V transposed)

  // Phase A: shifted-window gather, f32 -> f16 into LDS
  {
    int t  = tid >> 2;             // token 0..63 (4 threads per token)
    int c0 = (tid & 3) * 64;
    int tr = t >> 3, tc = t & 7;
    int sh = (wi * 8 + tr + SHIFT_) & 255;
    int sw = (wj * 8 + tc + SHIFT_) & 255;
    const float4* src =
        (const float4*)(x + (((size_t)wb * HW_ + sh) * HW_ + sw) * DIM_ + c0);
    half_t* dst = xs + t * XP + c0;
#pragma unroll
    for (int u = 0; u < 16; ++u) {
      float4 f = src[u];
      v4h hv; hv[0] = (half_t)f.x; hv[1] = (half_t)f.y;
      hv[2] = (half_t)f.z; hv[3] = (half_t)f.w;
      *(v4h*)(dst + u * 4) = hv;
    }
  }
  __syncthreads();

  // Phase B: QKV projection for this head: [64,256] @ W_head^T -> q,k,v [64,32]
#pragma unroll
  for (int o = 0; o < 3; ++o) {
    const int rowbase = o * DIM_ + h * HD_;
#pragma unroll
    for (int j = 0; j < 2; ++j) {
      const int col = 16 * j + l15;
      const half_t* wB = wq16 + (size_t)(rowbase + col) * DIM_;
      const float bb = qkv_b[rowbase + col];
#pragma unroll
      for (int i = 0; i < 4; ++i) {
        v8f acc = {};
        const half_t* aRow = xs + (16 * i + l15) * XP;
        for (int kk = 0; kk < 8; ++kk) {
          v16h a = loadA(aRow + 32 * kk, g);
          v16h b = loadB(wB + 32 * kk, g);
          acc = wmma_f16(a, b, acc);
        }
        if (o == 0) {
#pragma unroll
          for (int r = 0; r < 8; ++r)
            qs[(16 * i + 8 * g + r) * QP + col] = (half_t)((acc[r] + bb) * SCALE_);
        } else if (o == 1) {
#pragma unroll
          for (int r = 0; r < 8; ++r)
            ks[(16 * i + 8 * g + r) * QP + col] = (half_t)(acc[r] + bb);
        } else {
          v8h hv;
#pragma unroll
          for (int r = 0; r < 8; ++r) hv[r] = (half_t)(acc[r] + bb);
          *(v8h*)(vt + col * VP + 16 * i + 8 * g) = hv;   // transposed store
        }
      }
    }
  }
  __syncthreads();   // everyone done with xs -> safe to alias proj input over it

  // Phase C: S = q @ k^T  (16 tiles, K=32 exact), register resident
  v8f S[4][4];
#pragma unroll
  for (int i = 0; i < 4; ++i) {
    v16h a = loadA(qs + (16 * i + l15) * QP, g);
#pragma unroll
    for (int j = 0; j < 4; ++j) {
      v16h b = loadB(ks + (16 * j + l15) * QP, g);
      v8f z = {};
      S[i][j] = wmma_f16(a, b, z);
    }
  }

  // relative-position bias (pre-gathered [h][m][n])
  const float* bh = bfull + h * 4096;
#pragma unroll
  for (int i = 0; i < 4; ++i)
#pragma unroll
    for (int j = 0; j < 4; ++j)
#pragma unroll
      for (int r = 0; r < 8; ++r)
        S[i][j][r] += bh[(16 * i + 8 * g + r) * 64 + 16 * j + l15];

  // row softmax: rows live across the 16 lanes of each half-wave group
#pragma unroll
  for (int i = 0; i < 4; ++i) {
#pragma unroll
    for (int r = 0; r < 8; ++r) {
      float mx = fmaxf(fmaxf(S[i][0][r], S[i][1][r]),
                       fmaxf(S[i][2][r], S[i][3][r]));
      mx = fmaxf(mx, __shfl_xor(mx, 1, 32));
      mx = fmaxf(mx, __shfl_xor(mx, 2, 32));
      mx = fmaxf(mx, __shfl_xor(mx, 4, 32));
      mx = fmaxf(mx, __shfl_xor(mx, 8, 32));
      float sum = 0.f;
#pragma unroll
      for (int j = 0; j < 4; ++j) {
        float e = __expf(S[i][j][r] - mx);
        S[i][j][r] = e;
        sum += e;
      }
      sum += __shfl_xor(sum, 1, 32);
      sum += __shfl_xor(sum, 2, 32);
      sum += __shfl_xor(sum, 4, 32);
      sum += __shfl_xor(sum, 8, 32);
      float inv = 1.0f / sum;
#pragma unroll
      for (int j = 0; j < 4; ++j) S[i][j][r] *= inv;
    }
  }

  // P -> LDS as f16 (overwrites this head's q/k; same-wave LDS is in-order)
#pragma unroll
  for (int i = 0; i < 4; ++i)
#pragma unroll
    for (int j = 0; j < 4; ++j)
#pragma unroll
      for (int r = 0; r < 8; ++r)
        ps[(16 * i + 8 * g + r) * PP + 16 * j + l15] = (half_t)S[i][j][r];

  // out_h = P @ V : [64,64] @ [64,32]
  half_t* po = xs;   // [64][XP] proj input, all heads interleaved by channel
#pragma unroll
  for (int i = 0; i < 4; ++i) {
#pragma unroll
    for (int jn = 0; jn < 2; ++jn) {
      v8f acc = {};
#pragma unroll
      for (int kk = 0; kk < 2; ++kk) {
        v16h a = loadA(ps + (16 * i + l15) * PP + 32 * kk, g);
        v16h b = loadB(vt + (16 * jn + l15) * VP + 32 * kk, g);
        acc = wmma_f16(a, b, acc);
      }
#pragma unroll
      for (int r = 0; r < 8; ++r)
        po[(16 * i + 8 * g + r) * XP + h * HD_ + 16 * jn + l15] = (half_t)acc[r];
    }
  }
  __syncthreads();

  // Phase D: output projection [64,256]@[256,256] + reverse-shift scatter (f32)
#pragma unroll
  for (int jt = 0; jt < 2; ++jt) {
    const int j = 2 * h + jt;            // n-tile 0..15, 2 per wave
    const int col = 16 * j + l15;
    const half_t* wB = wp16 + (size_t)col * DIM_;
    const float pb = proj_b[col];
#pragma unroll
    for (int i = 0; i < 4; ++i) {
      v8f acc = {};
      const half_t* aRow = po + (16 * i + l15) * XP;
      for (int kk = 0; kk < 8; ++kk) {
        v16h a = loadA(aRow + 32 * kk, g);
        v16h b = loadB(wB + 32 * kk, g);
        acc = wmma_f16(a, b, acc);
      }
      const int tr = 2 * i + g;                       // token row in window
      const int gh = (wi * 8 + tr + SHIFT_) & 255;
#pragma unroll
      for (int r = 0; r < 8; ++r) {                   // token col in window = r
        int gw = (wj * 8 + r + SHIFT_) & 255;
        out[(((size_t)wb * HW_ + gh) * HW_ + gw) * DIM_ + col] = acc[r] + pb;
      }
    }
  }
}

extern "C" void kernel_launch(void* const* d_in, const int* in_sizes, int n_in,
                              void* d_out, int out_size, void* d_ws, size_t ws_size,
                              hipStream_t stream) {
  (void)in_sizes; (void)n_in; (void)out_size; (void)ws_size;
  const float* x      = (const float*)d_in[0];
  const float* qkv_w  = (const float*)d_in[1];
  const float* qkv_b  = (const float*)d_in[2];
  const float* proj_w = (const float*)d_in[3];
  const float* proj_b = (const float*)d_in[4];
  const float* btab   = (const float*)d_in[5];
  const int*   ridx   = (const int*)d_in[6];
  float* out = (float*)d_out;

  // workspace: f16 qkv weights | f16 proj weights | pre-gathered bias (f32)
  half_t* wq16 = (half_t*)d_ws;                 // 196608 halves
  half_t* wp16 = wq16 + 3 * DIM_ * DIM_;        //  65536 halves
  float*  bfull = (float*)(wp16 + DIM_ * DIM_); //  32768 floats  (~656 KB total)

  swin_prepack<<<768, 256, 0, stream>>>(qkv_w, proj_w, btab, ridx, wq16, wp16, bfull);
  swin_attn<<<NWIN, 256, 0, stream>>>(x, qkv_b, proj_b, wq16, wp16, bfull, out);
}
